// CustomModel_32564442038771
// MI455X (gfx1250) — matrix operand order repaired and checked by reading, not compiled, not run on hardware
//
#include <hip/hip_runtime.h>
#include <hip/hip_bf16.h>

// ---------------------------------------------------------------------------
// CDNA5 (gfx1250) fused implementation of the hero-policy network.
// Every linear layer is a tall GEMM (M = 16K..160K rows) run as 16-row tiles
// through v_wmma_f32_16x16x32_bf16 (bf16 A/B, f32 accumulate).  GEMM-to-GEMM
// intermediates are bf16; LayerNorm / softmax / logits / argmax in fp32.
// bf16-sourced A-tiles are staged into LDS with GLOBAL_LOAD_ASYNC_TO_LDS_B128
// (ASYNCcnt-tracked, drained with s_wait_asynccnt before the barrier).
// ---------------------------------------------------------------------------

#define NB   16384      // batch
#define TOT  2357       // floats per input row

// input-field offsets (cumsum of SPLITS)
#define OFF_HERO   0
#define OFF_MPF    1
#define OFF_MPB    65
#define OFF_ALF    225
#define OFF_ALB    481
#define OFF_ENF    1121
#define OFF_ENB    1441
#define OFF_SKF    2241
#define OFF_AMSK   2337
#define OFF_EMSK   2342
#define OFF_SMSK   2347
#define OFF_STA    2350
#define OFF_STE    2353

typedef __attribute__((ext_vector_type(16))) __bf16 bf16x16;
typedef __attribute__((ext_vector_type(8)))  float  v8f;

__device__ __forceinline__ float gelu_f(float x) {
  return 0.5f * x * (1.0f + erff(x * 0.7071067811865475f));
}

// ---- gfx1250 async global->LDS copy (16 bytes per lane) --------------------
__device__ __forceinline__ void cp_async16(void* ldst, const void* gsrc) {
  unsigned lds_off =
      (unsigned)(unsigned long long)(__attribute__((address_space(3))) char*)(char*)ldst;
  asm volatile("global_load_async_to_lds_b128 %0, %1, off"
               :
               : "v"(lds_off), "v"((unsigned long long)gsrc)
               : "memory");
}
__device__ __forceinline__ void cp_async_wait() {
  asm volatile("s_wait_asynccnt 0" ::: "memory");
}

// A fragment: 16x32 bf16 tile staged row-major in LDS (lda = padded K).
// Contiguous k order used here (not the ISA table order): lanes 0-15 -> row M=lane, K in {0..15};
//             lanes16-31 -> row M=lane-16, K in {16..31}.
__device__ __forceinline__ bf16x16 a_frag(const __bf16* A, int lda, int k0, int lane) {
  const int m = lane & 15, half = lane >> 4;
  const __bf16* row = A + m * lda + k0 + half * 16;
  bf16x16 f;
#pragma unroll
  for (int i = 0; i < 16; ++i) f[i] = row[i];
  return f;
}

// B fragment straight from global bf16 weights (row-major K x N).
// Contiguous k order as in a_frag (not the ISA table order): lane holds column N = lane&15; lanes 0-15 hold K = k0..k0+15,
// lanes 16-31 hold K = k0+16..k0+31 (2 values per VGPR).
__device__ __forceinline__ bf16x16 b_frag(const __bf16* W, int ldw, int k0, int n0, int lane) {
  const int n = lane & 15, half = lane >> 4;
  const __bf16* p = W + (size_t)(k0 + half * 16) * ldw + n0 + n;
  bf16x16 f;
#pragma unroll
  for (int i = 0; i < 16; ++i) f[i] = p[(size_t)i * ldw];
  return f;
}

// 16-row x 16-col output strip over the full K dimension (KSTEPS * 32).
template <int KSTEPS>
__device__ __forceinline__ v8f wmma_strip(const __bf16* A, int lda,
                                          const __bf16* W, int ldw, int n0, int lane) {
  v8f acc = {0.f, 0.f, 0.f, 0.f, 0.f, 0.f, 0.f, 0.f};
  for (int ks = 0; ks < KSTEPS; ++ks) {
    bf16x16 a = a_frag(A, lda, ks * 32, lane);
    bf16x16 b = b_frag(W, ldw, ks * 32, n0, lane);
    acc = __builtin_amdgcn_wmma_f32_16x16x32_bf16(false, a, false, b, (short)0, acc,
                                                  false, false);
  }
  return acc;
}

// ---------------------------------------------------------------------------
// weight conversion
// ---------------------------------------------------------------------------
__global__ void k_cvt(const float* __restrict__ s, __bf16* __restrict__ d, int n) {
  int i = blockIdx.x * 256 + threadIdx.x;
  if (i < n) d[i] = (__bf16)s[i];
}

// W_buff (16x64) -> 32x64 with zero-padded K rows 16..31
__global__ void k_cvt_buffpad(const float* __restrict__ s, __bf16* __restrict__ d) {
  int i = blockIdx.x * 256 + threadIdx.x;
  if (i < 32 * 64) {
    int r = i >> 6, c = i & 63;
    d[i] = (r < 16) ? (__bf16)s[r * 64 + c] : (__bf16)0.f;
  }
}

// ---------------------------------------------------------------------------
// buff MLP: for each (sample, unit u in 0..9) take max over 10 buffs of
// GELU(buff16 @ W_buff + b). K=16 padded to 32 (weight rows zeroed).
// grid (NB/16, 10), block 128.
// ---------------------------------------------------------------------------
__global__ __launch_bounds__(128) void k_buff(const float* __restrict__ x,
                                              const __bf16* __restrict__ wb,
                                              const float* __restrict__ bias,
                                              __bf16* __restrict__ bmax) {
  __shared__ __align__(16) __bf16 As[16 * 32];
  const int s0 = blockIdx.x * 16, u = blockIdx.y;
  const int tid = threadIdx.x, lane = tid & 31, w = tid >> 5;
  for (int i = tid; i < 16 * 32; i += 128) As[i] = (__bf16)0.f;  // zero K pad
  const int base = (u == 0) ? OFF_MPB
                 : (u < 5)  ? OFF_ALB + (u - 1) * 160
                            : OFF_ENB + (u - 5) * 160;
  const int n0 = w * 16, nn = lane & 15, half = lane >> 4;
  float mx[8];
#pragma unroll
  for (int i = 0; i < 8; ++i) mx[i] = -3.0e38f;
  for (int b = 0; b < 10; ++b) {
    __syncthreads();
    for (int i = tid; i < 256; i += 128) {
      int m = i >> 4, c = i & 15;
      As[m * 32 + c] = (__bf16)x[(size_t)(s0 + m) * TOT + base + b * 16 + c];
    }
    __syncthreads();
    bf16x16 a = a_frag(As, 32, 0, lane);
    bf16x16 bb = b_frag(wb, 64, 0, n0, lane);
    v8f acc = {0.f, 0.f, 0.f, 0.f, 0.f, 0.f, 0.f, 0.f};
    acc = __builtin_amdgcn_wmma_f32_16x16x32_bf16(false, a, false, bb, (short)0, acc,
                                                  false, false);
#pragma unroll
    for (int i = 0; i < 8; ++i) mx[i] = fmaxf(mx[i], gelu_f(acc[i] + bias[n0 + nn]));
  }
#pragma unroll
  for (int i = 0; i < 8; ++i) {
    int m = i + 8 * half;
    bmax[((size_t)(s0 + m) * 10 + u) * 64 + n0 + nn] = (__bf16)mx[i];
  }
}

// ---------------------------------------------------------------------------
// mp encoder GEMM: A = [mp_feat(64) | hero_emb(64) | buffmax_u0(64)], K=192,
// Y = GELU(A@W_mp + b)  (f32, pre-LN).  grid NB/16, block 128.
// ---------------------------------------------------------------------------
__global__ __launch_bounds__(128) void k_mp(const float* __restrict__ x,
                                            const float* __restrict__ hero_emb,
                                            const __bf16* __restrict__ bmax,
                                            const __bf16* __restrict__ wb,
                                            const float* __restrict__ bias,
                                            float* __restrict__ Y) {
  __shared__ __align__(16) __bf16 As[16 * 192];
  const int s0 = blockIdx.x * 16, tid = threadIdx.x, lane = tid & 31, w = tid >> 5;
  for (int i = tid; i < 16 * 192; i += 128) {
    int m = i / 192, c = i % 192, s = s0 + m;
    if (c < 64) {
      As[i] = (__bf16)x[(size_t)s * TOT + OFF_MPF + c];
    } else if (c < 128) {
      int hid = (int)x[(size_t)s * TOT + OFF_HERO];
      As[i] = (__bf16)hero_emb[hid * 64 + (c - 64)];
    } else {
      As[i] = bmax[(size_t)s * 640 + (c - 128)];
    }
  }
  __syncthreads();
  const int nn = lane & 15, half = lane >> 4;
  for (int t = 0; t < 4; ++t) {
    int n0 = (w * 4 + t) * 16;
    v8f acc = wmma_strip<6>(As, 192, wb, 256, n0, lane);
#pragma unroll
    for (int i = 0; i < 8; ++i) {
      int m = i + 8 * half;
      Y[(size_t)(s0 + m) * 256 + n0 + nn] = gelu_f(acc[i] + bias[n0 + nn]);
    }
  }
}

// ---------------------------------------------------------------------------
// ally/enemy encoder GEMM: rows r = s*ups + a, A = [feat64 | buffmax64], K=128.
// grid NB*ups/16, block 128.
// ---------------------------------------------------------------------------
__global__ __launch_bounds__(128) void k_unit(const float* __restrict__ x,
                                              const __bf16* __restrict__ bmax,
                                              const __bf16* __restrict__ wb,
                                              const float* __restrict__ bias,
                                              float* __restrict__ Y,
                                              int ups, int featOff, int ubase) {
  __shared__ __align__(16) __bf16 As[16 * 128];
  const int r0 = blockIdx.x * 16, tid = threadIdx.x, lane = tid & 31, w = tid >> 5;
  for (int i = tid; i < 16 * 128; i += 128) {
    int m = i >> 7, c = i & 127, r = r0 + m, s = r / ups, a = r % ups;
    if (c < 64)
      As[i] = (__bf16)x[(size_t)s * TOT + featOff + a * 64 + c];
    else
      As[i] = bmax[((size_t)s * 10 + ubase + a) * 64 + (c - 64)];
  }
  __syncthreads();
  const int nn = lane & 15, half = lane >> 4;
  for (int t = 0; t < 4; ++t) {
    int n0 = (w * 4 + t) * 16;
    v8f acc = wmma_strip<4>(As, 128, wb, 256, n0, lane);
#pragma unroll
    for (int i = 0; i < 8; ++i) {
      int m = i + 8 * half;
      Y[(size_t)(r0 + m) * 256 + n0 + nn] = gelu_f(acc[i] + bias[n0 + nn]);
    }
  }
}

// ---------------------------------------------------------------------------
// LayerNorm over (U units x 256 feats) jointly per sample + per-column max
// over units. grid NB, block 256.
// ---------------------------------------------------------------------------
template <int U>
__global__ __launch_bounds__(256) void k_ln(const float* __restrict__ Y,
                                            __bf16* __restrict__ H,
                                            __bf16* __restrict__ Hmax) {
  __shared__ float red[256];
  const int s = blockIdx.x, n = threadIdx.x;
  float v[U], sum = 0.f, sq = 0.f;
#pragma unroll
  for (int a = 0; a < U; ++a) {
    v[a] = Y[((size_t)s * U + a) * 256 + n];
    sum += v[a];
    sq += v[a] * v[a];
  }
  red[n] = sum;
  __syncthreads();
  for (int st = 128; st > 0; st >>= 1) {
    if (n < st) red[n] += red[n + st];
    __syncthreads();
  }
  float tot = red[0];
  __syncthreads();
  red[n] = sq;
  __syncthreads();
  for (int st = 128; st > 0; st >>= 1) {
    if (n < st) red[n] += red[n + st];
    __syncthreads();
  }
  float tot2 = red[0];
  const float inv = 1.0f / (256.0f * U);
  float mean = tot * inv;
  float var = tot2 * inv - mean * mean;
  float rs = rsqrtf(var + 1e-5f);
  float mx = -3.0e38f;
#pragma unroll
  for (int a = 0; a < U; ++a) {
    float h = (v[a] - mean) * rs;
    H[((size_t)s * U + a) * 256 + n] = (__bf16)h;
    mx = fmaxf(mx, h);
  }
  if (Hmax) Hmax[(size_t)s * 256 + n] = (__bf16)mx;
}

// ---------------------------------------------------------------------------
// fc: A = concat_h (768 bf16) staged via async-to-LDS, K=24 steps.
// Fused epilogue: LN(256), skill logits (*mask + log eps), argmax ->
// slot_pred, values, sta/ste gather.  grid NB/16, block 128.
// ---------------------------------------------------------------------------
__global__ __launch_bounds__(128) void k_fc(const __bf16* __restrict__ mp_h,
                                            const __bf16* __restrict__ amax,
                                            const __bf16* __restrict__ emax,
                                            const __bf16* __restrict__ wb,
                                            const float* __restrict__ bias,
                                            const float* __restrict__ x,
                                            const float* __restrict__ Wsk,
                                            const float* __restrict__ bsk,
                                            const float* __restrict__ Wval,
                                            const float* __restrict__ bval,
                                            float* __restrict__ out,
                                            int* __restrict__ slotpred) {
  __shared__ __align__(16) __bf16 As[16 * 768];
  __shared__ float Z[16 * 256];
  const int s0 = blockIdx.x * 16, tid = threadIdx.x, lane = tid & 31, w = tid >> 5;
  __builtin_prefetch(wb + (size_t)tid * 256, 0, 2);  // warm W_fc stream (global_prefetch_b8)
  // 16 rows x 96 16-byte chunks, async global->LDS
  for (int c = tid; c < 16 * 96; c += 128) {
    int m = c / 96, col = (c % 96) * 8, s = s0 + m;
    const __bf16* src = (col < 256) ? mp_h + (size_t)s * 256 + col
                      : (col < 512) ? amax + (size_t)s * 256 + (col - 256)
                                    : emax + (size_t)s * 256 + (col - 512);
    cp_async16(&As[m * 768 + col], src);
  }
  cp_async_wait();
  __syncthreads();
  const int nn = lane & 15, half = lane >> 4;
  for (int t = 0; t < 4; ++t) {
    int n0 = (w * 4 + t) * 16;
    v8f acc = wmma_strip<24>(As, 768, wb, 256, n0, lane);
#pragma unroll
    for (int i = 0; i < 8; ++i) {
      int m = i + 8 * half;
      Z[m * 256 + n0 + nn] = gelu_f(acc[i] + bias[n0 + nn]);
    }
  }
  __syncthreads();
  if (tid < 16) {
    const int m = tid, s = s0 + m;
    float sum = 0.f, sq = 0.f;
    for (int n = 0; n < 256; ++n) {
      float z = Z[m * 256 + n];
      sum += z;
      sq += z * z;
    }
    float mean = sum * (1.0f / 256.0f);
    float var = sq * (1.0f / 256.0f) - mean * mean;
    float rs = rsqrtf(var + 1e-5f);
    float lg0 = bsk[0], lg1 = bsk[1], lg2 = bsk[2], val = bval[0];
    for (int n = 0; n < 256; ++n) {
      float h = (Z[m * 256 + n] - mean) * rs;
      lg0 += h * Wsk[n * 3 + 0];
      lg1 += h * Wsk[n * 3 + 1];
      lg2 += h * Wsk[n * 3 + 2];
      val += h * Wval[n];
    }
    const float leps = -16.11809565095832f;  // log(1e-7)
    float lg[3] = {lg0, lg1, lg2};
    int best = 0;
    float bv = -3.0e38f;
    for (int j = 0; j < 3; ++j) {
      float msk = x[(size_t)s * TOT + OFF_SMSK + j];
      float lo = lg[j] * msk + leps;
      out[(size_t)s * 13 + j] = lo;
      if (lo > bv) { bv = lo; best = j; }
    }
    slotpred[s] = best;
    out[(size_t)NB * 13 + s] = val;                                   // values
    out[(size_t)NB * 14 + s] = x[(size_t)s * TOT + OFF_STA + best];   // stami
    out[(size_t)NB * 15 + s] = x[(size_t)s * TOT + OFF_STE + best];   // stemi
  }
}

// ---------------------------------------------------------------------------
// attention keys: rows r = s*10 + u (u0=mp, u1-4=ally, u5-9=enemy), K=256,
// k = A@W_k + b (linear only).  A staged via async-to-LDS.
// grid NB*10/16, block 128.
// ---------------------------------------------------------------------------
__global__ __launch_bounds__(128) void k_keys(const __bf16* __restrict__ mp_h,
                                              const __bf16* __restrict__ ally_h,
                                              const __bf16* __restrict__ enemy_h,
                                              const __bf16* __restrict__ wb,
                                              const float* __restrict__ bias,
                                              __bf16* __restrict__ kall) {
  __shared__ __align__(16) __bf16 As[16 * 256];
  const int r0 = blockIdx.x * 16, tid = threadIdx.x, lane = tid & 31, w = tid >> 5;
  for (int c = tid; c < 16 * 32; c += 128) {
    int m = c >> 5, j = (c & 31) * 8, r = r0 + m, s = r / 10, u = r % 10;
    const __bf16* src = (u == 0) ? mp_h + (size_t)s * 256
                      : (u < 5)  ? ally_h + ((size_t)s * 4 + (u - 1)) * 256
                                 : enemy_h + ((size_t)s * 5 + (u - 5)) * 256;
    cp_async16(&As[m * 256 + j], src + j);
  }
  cp_async_wait();
  __syncthreads();
  const int nn = lane & 15, half = lane >> 4;
  for (int t = 0; t < 4; ++t) {
    int n0 = (w * 4 + t) * 16;
    v8f acc = wmma_strip<8>(As, 256, wb, 256, n0, lane);
#pragma unroll
    for (int i = 0; i < 8; ++i) {
      int m = i + 8 * half;
      kall[(size_t)(r0 + m) * 256 + n0 + nn] = (__bf16)(acc[i] + bias[n0 + nn]);
    }
  }
}

// ---------------------------------------------------------------------------
// gather + sb (128->128) + si (32->64), both GELU.  grid NB/16, block 128.
// ---------------------------------------------------------------------------
__global__ __launch_bounds__(128) void k_sbsi(const float* __restrict__ x,
                                              const float* __restrict__ semb,
                                              const int* __restrict__ slotpred,
                                              const __bf16* __restrict__ wsb,
                                              const float* __restrict__ bsb,
                                              const __bf16* __restrict__ wsi,
                                              const float* __restrict__ bsi,
                                              __bf16* __restrict__ sb,
                                              __bf16* __restrict__ si) {
  __shared__ __align__(16) __bf16 Ab[16 * 128];
  __shared__ __align__(16) __bf16 Ai[16 * 32];
  const int s0 = blockIdx.x * 16, tid = threadIdx.x, lane = tid & 31, w = tid >> 5;
  for (int i = tid; i < 16 * 128; i += 128) {
    int m = i >> 7, c = i & 127, s = s0 + m;
    int hid = (int)x[(size_t)s * TOT + OFF_HERO];
    int sid = slotpred[s] + 3 * hid;
    Ab[i] = (__bf16)semb[(size_t)sid * 128 + c];
  }
  for (int i = tid; i < 16 * 32; i += 128) {
    int m = i >> 5, c = i & 31, s = s0 + m;
    Ai[i] = (__bf16)x[(size_t)s * TOT + OFF_SKF + slotpred[s] * 32 + c];
  }
  __syncthreads();
  const int nn = lane & 15, half = lane >> 4;
  for (int t = 0; t < 2; ++t) {  // sb: N=128 -> 8 tiles, 2 per wave
    int n0 = (w * 2 + t) * 16;
    v8f acc = wmma_strip<4>(Ab, 128, wsb, 128, n0, lane);
#pragma unroll
    for (int i = 0; i < 8; ++i) {
      int m = i + 8 * half;
      sb[(size_t)(s0 + m) * 128 + n0 + nn] = (__bf16)gelu_f(acc[i] + bsb[n0 + nn]);
    }
  }
  {  // si: N=64 -> 4 tiles, 1 per wave
    int n0 = w * 16;
    v8f acc = wmma_strip<1>(Ai, 32, wsi, 64, n0, lane);
#pragma unroll
    for (int i = 0; i < 8; ++i) {
      int m = i + 8 * half;
      si[(size_t)(s0 + m) * 64 + n0 + nn] = (__bf16)gelu_f(acc[i] + bsi[n0 + nn]);
    }
  }
}

// ---------------------------------------------------------------------------
// qm: A = [concat_h(768) | sb(128) | si(64)] staged via async-to-LDS,
// K=30 steps, GELU -> merge bf16.  grid NB/16, block 128.
// ---------------------------------------------------------------------------
__global__ __launch_bounds__(128) void k_qm(const __bf16* __restrict__ mp_h,
                                            const __bf16* __restrict__ amax,
                                            const __bf16* __restrict__ emax,
                                            const __bf16* __restrict__ sb,
                                            const __bf16* __restrict__ si,
                                            const __bf16* __restrict__ wb,
                                            const float* __restrict__ bias,
                                            __bf16* __restrict__ merge) {
  __shared__ __align__(16) __bf16 As[16 * 960];
  const int s0 = blockIdx.x * 16, tid = threadIdx.x, lane = tid & 31, w = tid >> 5;
  __builtin_prefetch(wb + (size_t)tid * 256, 0, 2);  // warm W_qm stream
  for (int c = tid; c < 16 * 120; c += 128) {
    int m = c / 120, col = (c % 120) * 8, s = s0 + m;
    const __bf16* src = (col < 256) ? mp_h + (size_t)s * 256 + col
                      : (col < 512) ? amax + (size_t)s * 256 + (col - 256)
                      : (col < 768) ? emax + (size_t)s * 256 + (col - 512)
                      : (col < 896) ? sb + (size_t)s * 128 + (col - 768)
                                    : si + (size_t)s * 64 + (col - 896);
    cp_async16(&As[m * 960 + col], src);
  }
  cp_async_wait();
  __syncthreads();
  const int nn = lane & 15, half = lane >> 4;
  for (int t = 0; t < 4; ++t) {
    int n0 = (w * 4 + t) * 16;
    v8f acc = wmma_strip<30>(As, 960, wb, 256, n0, lane);
#pragma unroll
    for (int i = 0; i < 8; ++i) {
      int m = i + 8 * half;
      merge[(size_t)(s0 + m) * 256 + n0 + nn] = (__bf16)gelu_f(acc[i] + bias[n0 + nn]);
    }
  }
}

// ---------------------------------------------------------------------------
// q = merge @ W_q + b (linear only), f32 out.  A staged via async-to-LDS.
// ---------------------------------------------------------------------------
__global__ __launch_bounds__(128) void k_q(const __bf16* __restrict__ merge,
                                           const __bf16* __restrict__ wb,
                                           const float* __restrict__ bias,
                                           float* __restrict__ qv) {
  __shared__ __align__(16) __bf16 As[16 * 256];
  const int s0 = blockIdx.x * 16, tid = threadIdx.x, lane = tid & 31, w = tid >> 5;
  for (int c = tid; c < 16 * 32; c += 128) {
    int m = c >> 5, j = (c & 31) * 8;
    cp_async16(&As[m * 256 + j], merge + (size_t)(s0 + m) * 256 + j);
  }
  cp_async_wait();
  __syncthreads();
  const int nn = lane & 15, half = lane >> 4;
  for (int t = 0; t < 4; ++t) {
    int n0 = (w * 4 + t) * 16;
    v8f acc = wmma_strip<8>(As, 256, wb, 256, n0, lane);
#pragma unroll
    for (int i = 0; i < 8; ++i) {
      int m = i + 8 * half;
      qv[(size_t)(s0 + m) * 256 + n0 + nn] = acc[i] + bias[n0 + nn];
    }
  }
}

// ---------------------------------------------------------------------------
// finale: dot(q,k)/16, softmax over units 0-4 (mp+ally) and 5-9 (enemy),
// *mask, exp -> cli[3..13].  grid NB/16, block 160 (thread = (sample, unit)).
// ---------------------------------------------------------------------------
__global__ __launch_bounds__(160) void k_final(const float* __restrict__ qv,
                                               const __bf16* __restrict__ kall,
                                               const float* __restrict__ x,
                                               float* __restrict__ out) {
  __shared__ float dots[160];
  const int s0 = blockIdx.x * 16, tid = threadIdx.x;
  const int sl = tid / 10, u = tid % 10, s = s0 + sl;
  const float* q = qv + (size_t)s * 256;
  const __bf16* kk = kall + ((size_t)s * 10 + u) * 256;
  float d = 0.f;
  for (int n = 0; n < 256; ++n) d += q[n] * (float)kk[n];
  dots[tid] = d * (1.0f / 16.0f);  // / sqrt(ATT_H=256)
  __syncthreads();
  if (u == 0) {
    const float* dd = &dots[sl * 10];
    for (int g = 0; g < 2; ++g) {
      float m = -3.0e38f;
      for (int j = 0; j < 5; ++j) m = fmaxf(m, dd[g * 5 + j]);
      float e[5], sum = 0.f;
      for (int j = 0; j < 5; ++j) { e[j] = expf(dd[g * 5 + j] - m); sum += e[j]; }
      float isum = 1.0f / sum;
      for (int j = 0; j < 5; ++j) {
        float msk = x[(size_t)s * TOT + (g == 0 ? OFF_AMSK : OFF_EMSK) + j];
        out[(size_t)s * 13 + 3 + g * 5 + j] = expf(e[j] * isum * msk);
      }
    }
  }
}

// ---------------------------------------------------------------------------
extern "C" void kernel_launch(void* const* d_in, const int* in_sizes, int n_in,
                              void* d_out, int out_size, void* d_ws, size_t ws_size,
                              hipStream_t stream) {
  (void)in_sizes; (void)n_in; (void)out_size; (void)ws_size;
  const float* x        = (const float*)d_in[0];
  const float* W_buff   = (const float*)d_in[1];
  const float* b_buff   = (const float*)d_in[2];
  const float* hero_emb = (const float*)d_in[3];
  const float* W_mp     = (const float*)d_in[4];
  const float* b_mp     = (const float*)d_in[5];
  const float* W_ally   = (const float*)d_in[6];
  const float* b_ally   = (const float*)d_in[7];
  const float* W_enemy  = (const float*)d_in[8];
  const float* b_enemy  = (const float*)d_in[9];
  const float* W_fc     = (const float*)d_in[10];
  const float* b_fc     = (const float*)d_in[11];
  const float* W_skill  = (const float*)d_in[12];
  const float* b_skill  = (const float*)d_in[13];
  const float* skill_emb= (const float*)d_in[14];
  const float* W_sb     = (const float*)d_in[15];
  const float* b_sb     = (const float*)d_in[16];
  const float* W_si     = (const float*)d_in[17];
  const float* b_si     = (const float*)d_in[18];
  const float* W_qm     = (const float*)d_in[19];
  const float* b_qm     = (const float*)d_in[20];
  const float* W_q      = (const float*)d_in[21];
  const float* b_q      = (const float*)d_in[22];
  const float* W_k      = (const float*)d_in[23];
  const float* b_k      = (const float*)d_in[24];
  const float* W_val    = (const float*)d_in[25];
  const float* b_val    = (const float*)d_in[26];
  float* out = (float*)d_out;

  // workspace carve-up (256B aligned)
  char* base = (char*)d_ws;
  size_t off = 0;
  auto alloc = [&](size_t bytes) -> void* {
    void* p = base + off;
    off = (off + bytes + 255) & ~(size_t)255;
    return p;
  };
  __bf16* wbBuff  = (__bf16*)alloc((size_t)32 * 64 * 2);
  __bf16* wbMp    = (__bf16*)alloc((size_t)192 * 256 * 2);
  __bf16* wbAlly  = (__bf16*)alloc((size_t)128 * 256 * 2);
  __bf16* wbEnemy = (__bf16*)alloc((size_t)128 * 256 * 2);
  __bf16* wbFc    = (__bf16*)alloc((size_t)768 * 256 * 2);
  __bf16* wbSb    = (__bf16*)alloc((size_t)128 * 128 * 2);
  __bf16* wbSi    = (__bf16*)alloc((size_t)32 * 64 * 2);
  __bf16* wbQm    = (__bf16*)alloc((size_t)960 * 256 * 2);
  __bf16* wbQ     = (__bf16*)alloc((size_t)256 * 256 * 2);
  __bf16* wbK     = (__bf16*)alloc((size_t)256 * 256 * 2);
  __bf16* bmax    = (__bf16*)alloc((size_t)NB * 640 * 2);
  float*  Yt      = (float*) alloc((size_t)NB * 5 * 256 * 4);  // reused mp/ally/enemy
  __bf16* mp_h    = (__bf16*)alloc((size_t)NB * 256 * 2);
  __bf16* ally_h  = (__bf16*)alloc((size_t)NB * 4 * 256 * 2);
  __bf16* enemy_h = (__bf16*)alloc((size_t)NB * 5 * 256 * 2);
  __bf16* amax    = (__bf16*)alloc((size_t)NB * 256 * 2);
  __bf16* emax    = (__bf16*)alloc((size_t)NB * 256 * 2);
  __bf16* kall    = (__bf16*)alloc((size_t)NB * 10 * 256 * 2);
  __bf16* sb      = (__bf16*)alloc((size_t)NB * 128 * 2);
  __bf16* si      = (__bf16*)alloc((size_t)NB * 64 * 2);
  __bf16* merge   = (__bf16*)alloc((size_t)NB * 256 * 2);
  float*  qv      = (float*) alloc((size_t)NB * 256 * 4);
  int*    slotp   = (int*)   alloc((size_t)NB * 4);

  auto cvt = [&](const float* s, __bf16* d, int n) {
    k_cvt<<<(n + 255) / 256, 256, 0, stream>>>(s, d, n);
  };
  k_cvt_buffpad<<<8, 256, 0, stream>>>(W_buff, wbBuff);
  cvt(W_mp, wbMp, 192 * 256);
  cvt(W_ally, wbAlly, 128 * 256);
  cvt(W_enemy, wbEnemy, 128 * 256);
  cvt(W_fc, wbFc, 768 * 256);
  cvt(W_sb, wbSb, 128 * 128);
  cvt(W_si, wbSi, 32 * 64);
  cvt(W_qm, wbQm, 960 * 256);
  cvt(W_q, wbQ, 256 * 256);
  cvt(W_k, wbK, 256 * 256);

  k_buff<<<dim3(NB / 16, 10), 128, 0, stream>>>(x, wbBuff, b_buff, bmax);

  k_mp<<<NB / 16, 128, 0, stream>>>(x, hero_emb, bmax, wbMp, b_mp, Yt);
  k_ln<1><<<NB, 256, 0, stream>>>(Yt, mp_h, (__bf16*)nullptr);

  k_unit<<<NB * 4 / 16, 128, 0, stream>>>(x, bmax, wbAlly, b_ally, Yt, 4, OFF_ALF, 1);
  k_ln<4><<<NB, 256, 0, stream>>>(Yt, ally_h, amax);

  k_unit<<<NB * 5 / 16, 128, 0, stream>>>(x, bmax, wbEnemy, b_enemy, Yt, 5, OFF_ENF, 5);
  k_ln<5><<<NB, 256, 0, stream>>>(Yt, enemy_h, emax);

  k_fc<<<NB / 16, 128, 0, stream>>>(mp_h, amax, emax, wbFc, b_fc, x, W_skill, b_skill,
                                    W_val, b_val, out, slotp);

  k_keys<<<NB * 10 / 16, 128, 0, stream>>>(mp_h, ally_h, enemy_h, wbK, b_k, kall);

  k_sbsi<<<NB / 16, 128, 0, stream>>>(x, skill_emb, slotp, wbSb, b_sb, wbSi, b_si, sb, si);

  k_qm<<<NB / 16, 128, 0, stream>>>(mp_h, amax, emax, sb, si, wbQm, b_qm, merge);

  k_q<<<NB / 16, 128, 0, stream>>>(merge, wbQ, b_q, qv);

  k_final<<<NB / 16, 160, 0, stream>>>(qv, kall, x, out);
}
